// PPI_66597762891843
// MI455X (gfx1250) — matrix operand order repaired and checked by reading, not compiled, not run on hardware
//
#include <hip/hip_runtime.h>
#include <hip/hip_bf16.h>

// CDNA5 / gfx1250, wave32. bf16 WMMA 16x16x32 with f32 accumulation.
// Recurrences: weights resident in LDS, async (ASYNCcnt) double-buffered
// embedding tile staging overlapping the per-step WMMA work.
typedef __attribute__((ext_vector_type(16))) __bf16 v16bf;
typedef __attribute__((ext_vector_type(8)))  float  v8f;

#define L_SEQ   512
#define BATCH   64
#define DIM     128
#define HID     128
#define RSTRIDE 136      // padded row stride (bf16 elems) for emb/h stage regions

union Frag { uint4 q[2]; v16bf v; };

static __device__ __forceinline__ unsigned short f2bf(float f) {
    unsigned int u = __float_as_uint(f);
    u += 0x7FFFu + ((u >> 16) & 1u);          // round-to-nearest-even
    return (unsigned short)(u >> 16);
}
static __device__ __forceinline__ unsigned int pack2(unsigned short lo, unsigned short hi) {
    return (unsigned int)lo | ((unsigned int)hi << 16);
}
static __device__ __forceinline__ float sigmoidf_(float x) { return 1.0f / (1.0f + __expf(-x)); }

// Async copy of 16B from global to LDS (GV mode), tracked by ASYNCcnt.
static __device__ __forceinline__ void async_b128(unsigned lds_off, const void* gptr) {
    asm volatile("global_load_async_to_lds_b128 %0, %1, off"
                 :: "v"(lds_off), "v"((unsigned long long)(uintptr_t)gptr)
                 : "memory");
}
static __device__ __forceinline__ void wait_async0() {
    asm volatile("s_wait_asynccnt 0x0" ::: "memory");
}

// ---------------------------------------------------------------------------
// Kernel 1: embedding gather + f32 -> bf16. Layout embbf[(t*64+b)*128 + d].
// ---------------------------------------------------------------------------
__global__ void gather_embed_bf16(const int* __restrict__ x,
                                  const float* __restrict__ embed,
                                  unsigned short* __restrict__ embbf) {
    int gid   = blockIdx.x * blockDim.x + threadIdx.x;   // [0, 32768*16)
    int row   = gid >> 4;                                // t*64 + b
    int chunk = gid & 15;                                // 8 elems per chunk
    int tok   = x[row];
    const float4* src = (const float4*)(embed + (size_t)tok * DIM + chunk * 8);
    float4 a = src[0], b = src[1];
    uint4 o;
    o.x = pack2(f2bf(a.x), f2bf(a.y));
    o.y = pack2(f2bf(a.z), f2bf(a.w));
    o.z = pack2(f2bf(b.x), f2bf(b.y));
    o.w = pack2(f2bf(b.z), f2bf(b.w));
    ((uint4*)embbf)[(size_t)row * 16 + chunk] = o;
}

// ---------------------------------------------------------------------------
// Kernel 2: build K=256 x N=512 bf16 weight matrices, pre-swizzled into the
// 16x16x32 B-fragment layout used here (contiguous k, same as A): frag[ks][ntile][lane][16 bf16], where
// lane l: N = ntile*16 + (l&15); half = l>>4; vgpr v holds K = ks*32 + 16*half + 2v{,+1}.
// which==0: LSTM [Wih_f^T ; Whh_f^T]    which==1: tree [Wiou|Wf ; Uiou|Uf]
// ---------------------------------------------------------------------------
__global__ void build_wfrags(const float* __restrict__ Wih, const float* __restrict__ Whh,
                             const float* __restrict__ Wiou, const float* __restrict__ Uiou,
                             const float* __restrict__ Wf,  const float* __restrict__ Uf,
                             unsigned short* __restrict__ fragL,
                             unsigned short* __restrict__ fragT) {
    int which = blockIdx.y;
    int g     = blockIdx.x * blockDim.x + threadIdx.x;   // [0, 8*32*32)
    int lane  = g & 31;
    int nt    = (g >> 5) & 31;
    int ks    = g >> 10;
    int n     = nt * 16 + (lane & 15);
    int half  = lane >> 4;
    unsigned short vals[16];
#pragma unroll
    for (int v = 0; v < 8; ++v) {
#pragma unroll
        for (int p = 0; p < 2; ++p) {
            int k = ks * 32 + 16 * half + 2 * v + p;     // 0..255
            float w;
            if (which == 0) {
                w = (k < 128) ? Wih[n * 128 + k] : Whh[n * 128 + (k - 128)];
            } else {
                if (n < 384) w = (k < 128) ? Wiou[k * 384 + n] : Uiou[(k - 128) * 384 + n];
                else         w = (k < 128) ? Wf[k * 128 + (n - 384)] : Uf[(k - 128) * 128 + (n - 384)];
            }
            vals[2 * v + p] = f2bf(w);
        }
    }
    uint4 q0, q1;
    q0.x = pack2(vals[0],  vals[1]);  q0.y = pack2(vals[2],  vals[3]);
    q0.z = pack2(vals[4],  vals[5]);  q0.w = pack2(vals[6],  vals[7]);
    q1.x = pack2(vals[8],  vals[9]);  q1.y = pack2(vals[10], vals[11]);
    q1.z = pack2(vals[12], vals[13]); q1.w = pack2(vals[14], vals[15]);
    uint4* dst = (uint4*)((which ? fragT : fragL) + (size_t)g * 16);
    dst[0] = q0; dst[1] = q1;
}

// ---------------------------------------------------------------------------
// Helpers for the recurrent kernels (32 waves, 1 persistent workgroup).
// Wave w: mtile = w>>3 (batch block), nt = w&7 (hidden block). Each wave owns
// the 4 gate tiles (nt, nt+8, nt+16, nt+24) -> cell update is fragment-local.
// ---------------------------------------------------------------------------
__device__ __forceinline__ void load_wlds(const unsigned short* __restrict__ wfrag,
                                          unsigned int* wlds, unsigned short* hA, int tid) {
    const uint4* src = (const uint4*)wfrag;              // 262144 B = 16384 uint4
    uint4* dst = (uint4*)wlds;
    for (int i = tid; i < 16384; i += 1024) dst[i] = src[i];
    uint4 z; z.x = z.y = z.z = z.w = 0u;
    uint4* az = (uint4*)hA;                              // 64*136*2 B = 1088 uint4
    for (int i = tid; i < 1088; i += 1024) az[i] = z;
}

__device__ __forceinline__ void step_gemm(const unsigned short* __restrict__ embcur,
                                          const unsigned short* __restrict__ hA,
                                          const unsigned int* wlds,
                                          int nt, int lane, int arow, int half,
                                          v8f& acc0, v8f& acc1, v8f& acc2, v8f& acc3) {
    const uint4* wq = (const uint4*)wlds;
#pragma unroll
    for (int ks = 0; ks < 8; ++ks) {
        Frag A;
        const unsigned short* ap = (ks < 4)
            ? (embcur + arow * RSTRIDE + ks * 32 + 16 * half)
            : (hA     + arow * RSTRIDE + (ks - 4) * 32 + 16 * half);
        A.q[0] = *(const uint4*)ap;
        A.q[1] = *(const uint4*)(ap + 8);
        int base = ((ks * 32 + nt) * 32 + lane) * 2;     // uint4 index; ntile stride = 64 uint4
        Frag B0, B1, B2, B3;
        B0.q[0] = wq[base];             B0.q[1] = wq[base + 1];
        B1.q[0] = wq[base + 8  * 64];   B1.q[1] = wq[base + 8  * 64 + 1];
        B2.q[0] = wq[base + 16 * 64];   B2.q[1] = wq[base + 16 * 64 + 1];
        B3.q[0] = wq[base + 24 * 64];   B3.q[1] = wq[base + 24 * 64 + 1];
        acc0 = __builtin_amdgcn_wmma_f32_16x16x32_bf16(false, A.v, false, B0.v, (short)0, acc0, false, false);
        acc1 = __builtin_amdgcn_wmma_f32_16x16x32_bf16(false, A.v, false, B1.v, (short)0, acc1, false, false);
        acc2 = __builtin_amdgcn_wmma_f32_16x16x32_bf16(false, A.v, false, B2.v, (short)0, acc2, false, false);
        acc3 = __builtin_amdgcn_wmma_f32_16x16x32_bf16(false, A.v, false, B3.v, (short)0, acc3, false, false);
    }
}

// ---------------------------------------------------------------------------
// Kernel 3: forward LSTM recurrence (length-masked), sent_vec = final h.
// Per step t: gates = [emb_t | h] @ [Wih^T;Whh^T] + b ; PyTorch order i,f,g,o.
// ---------------------------------------------------------------------------
__global__ __launch_bounds__(1024) void lstm_fwd(const unsigned short* __restrict__ embbf,
                                                 const unsigned short* __restrict__ wfrag,
                                                 const float* __restrict__ bias,
                                                 const int* __restrict__ lengths,
                                                 float* __restrict__ sent_vec) {
    __shared__ __align__(16) unsigned int   wlds[65536];              // 256 KB weights
    __shared__ __align__(16) unsigned short embA[2][BATCH * RSTRIDE]; // double-buffered emb tile
    __shared__ __align__(16) unsigned short hA[BATCH * RSTRIDE];      // h (K cols 128..255)
    const int tid  = threadIdx.x;
    const int lane = tid & 31, wave = tid >> 5;
    const int mtile = wave >> 3, nt = wave & 7;
    const int half = lane >> 4, mrow = lane & 15;
    const int hcol = nt * 16 + mrow;
    const int row = tid >> 4, chunk = tid & 15;          // staging: 16B per thread

    // async-stage tile 0 while weights stream into LDS
    async_b128((unsigned)(uintptr_t)&embA[0][row * RSTRIDE + chunk * 8],
               embbf + (size_t)row * DIM + chunk * 8);
    load_wlds(wfrag, wlds, hA, tid);

    const float bi = bias[hcol], bfv = bias[128 + hcol], bg = bias[256 + hcol], bo = bias[384 + hcol];
    int lenr[8];
#pragma unroll
    for (int r = 0; r < 8; ++r) lenr[r] = lengths[mtile * 16 + r + 8 * half];
    float creg[8], hreg[8];
#pragma unroll
    for (int r = 0; r < 8; ++r) { creg[r] = 0.f; hreg[r] = 0.f; }
    wait_async0();
    __syncthreads();

    for (int t = 0; t < L_SEQ; ++t) {
        if (t + 1 < L_SEQ)   // overlap next tile's HBM->LDS copy with this step's WMMAs
            async_b128((unsigned)(uintptr_t)&embA[(t + 1) & 1][row * RSTRIDE + chunk * 8],
                       embbf + ((size_t)(t + 1) * 64 + row) * DIM + chunk * 8);

        v8f acc0, acc1, acc2, acc3;
#pragma unroll
        for (int e = 0; e < 8; ++e) { acc0[e] = bi; acc1[e] = bfv; acc2[e] = bg; acc3[e] = bo; }
        step_gemm(embA[t & 1], hA, wlds, nt, lane, mtile * 16 + mrow, half, acc0, acc1, acc2, acc3);

#pragma unroll
        for (int r = 0; r < 8; ++r) {
            float iv = sigmoidf_(acc0[r]);
            float fv = sigmoidf_(acc1[r]);
            float gv = tanhf(acc2[r]);
            float ov = sigmoidf_(acc3[r]);
            float cn = fv * creg[r] + iv * gv;
            float hn = ov * tanhf(cn);
            if (t < lenr[r]) { creg[r] = cn; hreg[r] = hn; }   // length mask
        }
        __syncthreads();                                   // all LDS reads of this step done
#pragma unroll
        for (int r = 0; r < 8; ++r) {
            int b = mtile * 16 + r + 8 * half;
            hA[b * RSTRIDE + hcol] = f2bf(hreg[r]);        // publish h for step t+1
        }
        wait_async0();                                     // next emb tile landed in LDS
        __syncthreads();
    }
#pragma unroll
    for (int r = 0; r < 8; ++r) {
        int b = mtile * 16 + r + 8 * half;
        sent_vec[b * 128 + hcol] = hreg[r];
    }
}

// ---------------------------------------------------------------------------
// Kernel 4: tree-LSTM over the left-deep chain (h_j <- h_{j-1}, right child
// absent). Columns: [i(0:128) | o(128:256) | u(256:384) | f(384:512)].
// Root (c,h) captured at j == length-1.
// ---------------------------------------------------------------------------
__global__ __launch_bounds__(1024) void tree_fwd(const unsigned short* __restrict__ embbf,
                                                 const unsigned short* __restrict__ wfrag,
                                                 const float* __restrict__ biou,
                                                 const float* __restrict__ bfg,
                                                 const int* __restrict__ lengths,
                                                 float* __restrict__ t_state,
                                                 float* __restrict__ t_hidden) {
    __shared__ __align__(16) unsigned int   wlds[65536];
    __shared__ __align__(16) unsigned short embA[2][BATCH * RSTRIDE];
    __shared__ __align__(16) unsigned short hA[BATCH * RSTRIDE];
    const int tid  = threadIdx.x;
    const int lane = tid & 31, wave = tid >> 5;
    const int mtile = wave >> 3, nt = wave & 7;
    const int half = lane >> 4, mrow = lane & 15;
    const int hcol = nt * 16 + mrow;
    const int row = tid >> 4, chunk = tid & 15;

    async_b128((unsigned)(uintptr_t)&embA[0][row * RSTRIDE + chunk * 8],
               embbf + (size_t)row * DIM + chunk * 8);
    load_wlds(wfrag, wlds, hA, tid);

    const float bi = biou[hcol], bo = biou[128 + hcol], bu = biou[256 + hcol], bff = bfg[hcol];
    int lenr[8];
#pragma unroll
    for (int r = 0; r < 8; ++r) lenr[r] = lengths[mtile * 16 + r + 8 * half];
    float creg[8], hreg[8];
#pragma unroll
    for (int r = 0; r < 8; ++r) { creg[r] = 0.f; hreg[r] = 0.f; }
    wait_async0();
    __syncthreads();

    for (int j = 0; j < L_SEQ; ++j) {
        if (j + 1 < L_SEQ)
            async_b128((unsigned)(uintptr_t)&embA[(j + 1) & 1][row * RSTRIDE + chunk * 8],
                       embbf + ((size_t)(j + 1) * 64 + row) * DIM + chunk * 8);

        v8f acc0, acc1, acc2, acc3;
#pragma unroll
        for (int e = 0; e < 8; ++e) { acc0[e] = bi; acc1[e] = bo; acc2[e] = bu; acc3[e] = bff; }
        step_gemm(embA[j & 1], hA, wlds, nt, lane, mtile * 16 + mrow, half, acc0, acc1, acc2, acc3);

#pragma unroll
        for (int r = 0; r < 8; ++r) {
            int b  = mtile * 16 + r + 8 * half;
            float iv = sigmoidf_(acc0[r]);
            float ov = sigmoidf_(acc1[r]);
            float uv = tanhf(acc2[r]);
            float fv = sigmoidf_(acc3[r]);
            float cn = iv * uv + fv * creg[r];        // right child c == 0
            float hn = ov * tanhf(cn);
            creg[r] = cn; hreg[r] = hn;
            if (j == lenr[r] - 1) {                   // root of this sample's chain
                t_state[b * 128 + hcol]  = cn;
                t_hidden[b * 128 + hcol] = hn;
            }
        }
        __syncthreads();
#pragma unroll
        for (int r = 0; r < 8; ++r) {
            int b = mtile * 16 + r + 8 * half;
            hA[b * RSTRIDE + hcol] = f2bf(hreg[r]);
        }
        wait_async0();
        __syncthreads();
    }
}

// ---------------------------------------------------------------------------
// Kernel 5: classifier + BCE loss (scalar). 64 threads, one per sample.
// ---------------------------------------------------------------------------
__global__ void final_loss(const float* __restrict__ t_state, const float* __restrict__ t_hidden,
                           const float* __restrict__ sent_vec, const float* __restrict__ Wout,
                           const float* __restrict__ bout, const float* __restrict__ w1,
                           const float* __restrict__ w2, const int* __restrict__ y,
                           float* __restrict__ out) {
    __shared__ float red[64];
    int b = threadIdx.x;
    float a1 = *w1, a2 = *w2;
    float z0 = bout[0], z1 = bout[1];
    for (int k = 0; k < 128; ++k) {
        float ts = t_state[b * 128 + k];
        float th = a1 * t_hidden[b * 128 + k] + a2 * sent_vec[b * 128 + k];
        z0 += Wout[k] * ts + Wout[128 + k] * th;
        z1 += Wout[256 + k] * ts + Wout[384 + k] * th;
    }
    float p0 = sigmoidf_(z0), p1 = sigmoidf_(z1);
    int yy = y[b]; yy = yy < 0 ? 0 : (yy > 1 ? 1 : yy);
    float t0 = (yy == 0) ? 1.f : 0.f, t1 = 1.f - t0;
    float c0 = fminf(fmaxf(p0, 1e-7f), 1.f - 1e-7f);
    float c1 = fminf(fmaxf(p1, 1e-7f), 1.f - 1e-7f);
    float s = t0 * __logf(c0) + (1.f - t0) * __logf(1.f - c0)
            + t1 * __logf(c1) + (1.f - t1) * __logf(1.f - c1);
    red[b] = s;
    __syncthreads();
    if (b == 0) {
        float tot = 0.f;
        for (int i = 0; i < 64; ++i) tot += red[i];
        out[0] = -tot / 128.f;
    }
}

// ---------------------------------------------------------------------------
extern "C" void kernel_launch(void* const* d_in, const int* in_sizes, int n_in,
                              void* d_out, int out_size, void* d_ws, size_t ws_size,
                              hipStream_t stream) {
    const int*   x       = (const int*)d_in[0];
    const int*   y       = (const int*)d_in[1];
    const int*   lengths = (const int*)d_in[3];
    const float* embed   = (const float*)d_in[6];
    const float* Wih_f   = (const float*)d_in[7];
    const float* Whh_f   = (const float*)d_in[8];
    const float* b_f     = (const float*)d_in[9];
    const float* Wiou    = (const float*)d_in[13];
    const float* Uiou    = (const float*)d_in[14];
    const float* biou    = (const float*)d_in[15];
    const float* Wf      = (const float*)d_in[16];
    const float* Uf      = (const float*)d_in[17];
    const float* bf_     = (const float*)d_in[18];
    const float* Wout    = (const float*)d_in[19];
    const float* bout    = (const float*)d_in[20];
    const float* w1      = (const float*)d_in[21];
    const float* w2      = (const float*)d_in[22];
    // backward LSTM (d_in[10..12]) and p / child arrays are dead w.r.t. the loss.

    char* ws = (char*)d_ws;
    unsigned short* embbf = (unsigned short*)(ws);                 //  8,388,608 B
    unsigned short* fragL = (unsigned short*)(ws + 8388608);       //    262,144 B
    unsigned short* fragT = (unsigned short*)(ws + 8650752);       //    262,144 B
    float* sent   = (float*)(ws + 8912896);                        //     32,768 B
    float* tstate = (float*)(ws + 8945664);                        //     32,768 B
    float* thid   = (float*)(ws + 8978432);                        //     32,768 B

    gather_embed_bf16<<<2048, 256, 0, stream>>>(x, embed, embbf);
    dim3 gw(32, 2);
    build_wfrags<<<gw, 256, 0, stream>>>(Wih_f, Whh_f, Wiou, Uiou, Wf, Uf, fragL, fragT);
    lstm_fwd<<<1, 1024, 0, stream>>>(embbf, fragL, b_f, lengths, sent);
    tree_fwd<<<1, 1024, 0, stream>>>(embbf, fragT, biou, bf_, lengths, tstate, thid);
    final_loss<<<1, 64, 0, stream>>>(tstate, thid, sent, Wout, bout, w1, w2, y, (float*)d_out);
}
